// MultiScaleDynamicMaskHead_89670327205901
// MI455X (gfx1250) — matrix operand order repaired and checked by reading, not compiled, not run on hardware
//
#include <hip/hip_runtime.h>
#include <hip/hip_bf16.h>

// ---------------------------------------------------------------------------
// MultiScaleDynamicMaskHead for MI455X (gfx1250, wave32, WMMA).
// Pipeline:
//   0) transpose Wq/Wk/Wv -> f16 W^T (K-major) in ws
//   1) q  = LN(pro @ Wq^T + bq)            -> f16 (B,L,C)
//   2) k  = LN((f+p)@ Wk^T + bk)           -> f16 (B,C,N)  (transposed!)
//      v  = LN( f   @ Wv^T + bv)           -> f16 (B,N,C)
//   3) S  = q @ k^T                        -> f32 (B,L,N)
//   4) softmax over L + mask zero-fill     (in place)
//   5) O  = S @ v                          -> f32 (B,L,C)
//   6) out = relu(LN(O))                   -> f32 d_out
// All GEMMs use v_wmma_f32_16x16x32_f16 (f32 accumulate).
// ---------------------------------------------------------------------------

typedef __attribute__((ext_vector_type(16))) _Float16 v16h;
typedef __attribute__((ext_vector_type(8)))  _Float16 h8;
typedef __attribute__((ext_vector_type(8)))  float    v8f;

#define DEV __device__ __forceinline__

constexpr int Bb   = 2;
constexpr int L    = 512;
constexpr int C    = 256;
constexpr int NPIX = 160 * 160;   // 25600
constexpr float EPS = 1e-5f;

DEV v8f wmma_f16(v16h a, v16h b, v8f c) {
    return __builtin_amdgcn_wmma_f32_16x16x32_f16(
        /*neg_a=*/false, a, /*neg_b=*/false, b,
        /*c_mod=*/(short)0, c, /*reuse_a=*/false, /*reuse_b=*/false);
}

// ---------------------------------------------------------------------------
// Kernel 0: W (Cout,Cin) f32 -> W^T (Cin,Cout) f16   (B-operand friendly)
// ---------------------------------------------------------------------------
__global__ void __launch_bounds__(256)
wt_kernel(const float* __restrict__ W, _Float16* __restrict__ WT) {
    int idx = blockIdx.x * blockDim.x + threadIdx.x;   // 65536
    int c = idx >> 8;          // input channel (K)
    int j = idx & 255;         // output channel (N)
    WT[(size_t)c * C + j] = (_Float16)W[(size_t)j * C + c];
}

// ---------------------------------------------------------------------------
// Kernel 1: q projection + LN.  pro (B,L,C) row-major -> q16 (B,L,C) f16.
// One wave = one 16-row strip; block = 2 waves.
// ---------------------------------------------------------------------------
__global__ void __launch_bounds__(64)
proj_q_kernel(const float* __restrict__ pro, const _Float16* __restrict__ WT,
              const float* __restrict__ bias, const float* __restrict__ gamma,
              const float* __restrict__ beta, _Float16* __restrict__ out) {
    __shared__ float sOut[2][16][260];
    __shared__ float sMean[2][16], sRstd[2][16];
    const int lane = threadIdx.x & 31;
    const int wv   = threadIdx.x >> 5;
    const int strips = L / 32;
    const int b  = blockIdx.x / strips;
    const int m0 = (blockIdx.x % strips) * 32 + wv * 16;

    const int kb8 = (lane < 16) ? 0 : 8;
    const float* xr = pro + ((size_t)b * L + m0 + (lane & 15)) * C;

    v16h A[8];
    #pragma unroll
    for (int t = 0; t < 8; ++t) {
        #pragma unroll
        for (int i = 0; i < 8; ++i) {
            A[t][i]     = (_Float16)xr[t * 32 + kb8 + i];
            A[t][8 + i] = (_Float16)xr[t * 32 + kb8 + 16 + i];
        }
    }

    const int krow = (lane & 15) + ((lane < 16) ? 0 : 16);
    const int nn15 = lane & 15;
    const int mof  = (lane < 16) ? 0 : 8;
    for (int j0 = 0; j0 < C; j0 += 16) {
        float bb = bias[j0 + nn15];
        v8f acc;
        #pragma unroll
        for (int r = 0; r < 8; ++r) acc[r] = bb;
        #pragma unroll
        for (int t = 0; t < 8; ++t) {
            v16h Bf;
            #pragma unroll
            for (int i = 0; i < 8; ++i) {
                Bf[i]     = WT[(size_t)(t * 32 + kb8 + i) * C + j0 + nn15];
                Bf[8 + i] = WT[(size_t)(t * 32 + kb8 + 16 + i) * C + j0 + nn15];
            }
            acc = wmma_f16(A[t], Bf, acc);
        }
        #pragma unroll
        for (int r = 0; r < 8; ++r) sOut[wv][mof + r][j0 + nn15] = acc[r];
    }
    __syncthreads();
    if (lane < 16) {
        float s = 0.f;
        for (int c2 = 0; c2 < C; ++c2) s += sOut[wv][lane][c2];
        float m = s * (1.0f / C);
        float vv = 0.f;
        for (int c2 = 0; c2 < C; ++c2) { float d = sOut[wv][lane][c2] - m; vv += d * d; }
        sMean[wv][lane] = m;
        sRstd[wv][lane] = rsqrtf(vv * (1.0f / C) + EPS);
    }
    __syncthreads();
    for (int idx = lane; idx < 16 * C; idx += 32) {
        int col = idx & (C - 1), row = idx >> 8;
        float y = (sOut[wv][row][col] - sMean[wv][row]) * sRstd[wv][row] * gamma[col] + beta[col];
        out[((size_t)(b * L + m0 + row)) * C + col] = (_Float16)y;
    }
}

// ---------------------------------------------------------------------------
// Kernel 2: k / v projection + LN from channel-major features.
//   addpos=1 adds pos. transposed=1 writes (B,C,N), else (B,N,C).
// ---------------------------------------------------------------------------
__global__ void __launch_bounds__(64)
proj_feat_kernel(const float* __restrict__ f, const float* __restrict__ pos,
                 const _Float16* __restrict__ WT, const float* __restrict__ bias,
                 const float* __restrict__ gamma, const float* __restrict__ beta,
                 _Float16* __restrict__ out, int addpos, int transposed) {
    __shared__ float sOut[2][16][260];
    __shared__ float sMean[2][16], sRstd[2][16];
    const int lane = threadIdx.x & 31;
    const int wv   = threadIdx.x >> 5;
    const int strips = NPIX / 32;
    const int b  = blockIdx.x / strips;
    const int m0 = (blockIdx.x % strips) * 32 + wv * 16;

    const float* fb = f   + (size_t)b * C * NPIX;
    const float* pb = pos + (size_t)b * C * NPIX;
    const int mrow = m0 + (lane & 15);
    const int kb8  = (lane < 16) ? 0 : 8;

    v16h A[8];
    for (int t = 0; t < 8; ++t) {
        #pragma unroll
        for (int i = 0; i < 8; ++i) {
            int c0 = t * 32 + kb8 + i;
            int c1 = c0 + 16;
            float x0 = fb[(size_t)c0 * NPIX + mrow];   // coalesced across lanes 0-15
            float x1 = fb[(size_t)c1 * NPIX + mrow];
            if (addpos) {
                x0 += pb[(size_t)c0 * NPIX + mrow];
                x1 += pb[(size_t)c1 * NPIX + mrow];
            }
            A[t][i]     = (_Float16)x0;
            A[t][8 + i] = (_Float16)x1;
        }
    }

    const int krow = (lane & 15) + ((lane < 16) ? 0 : 16);
    const int nn15 = lane & 15;
    const int mof  = (lane < 16) ? 0 : 8;
    for (int j0 = 0; j0 < C; j0 += 16) {
        float bb = bias[j0 + nn15];
        v8f acc;
        #pragma unroll
        for (int r = 0; r < 8; ++r) acc[r] = bb;
        #pragma unroll
        for (int t = 0; t < 8; ++t) {
            v16h Bf;
            #pragma unroll
            for (int i = 0; i < 8; ++i) {
                Bf[i]     = WT[(size_t)(t * 32 + kb8 + i) * C + j0 + nn15];
                Bf[8 + i] = WT[(size_t)(t * 32 + kb8 + 16 + i) * C + j0 + nn15];
            }
            acc = wmma_f16(A[t], Bf, acc);
        }
        #pragma unroll
        for (int r = 0; r < 8; ++r) sOut[wv][mof + r][j0 + nn15] = acc[r];
    }
    __syncthreads();
    if (lane < 16) {
        float s = 0.f;
        for (int c2 = 0; c2 < C; ++c2) s += sOut[wv][lane][c2];
        float m = s * (1.0f / C);
        float vv = 0.f;
        for (int c2 = 0; c2 < C; ++c2) { float d = sOut[wv][lane][c2] - m; vv += d * d; }
        sMean[wv][lane] = m;
        sRstd[wv][lane] = rsqrtf(vv * (1.0f / C) + EPS);
    }
    __syncthreads();
    if (transposed) {
        for (int idx = lane; idx < 16 * C; idx += 32) {
            int row = idx & 15, col = idx >> 4;
            float y = (sOut[wv][row][col] - sMean[wv][row]) * sRstd[wv][row] * gamma[col] + beta[col];
            out[((size_t)b * C + col) * NPIX + (m0 + row)] = (_Float16)y;
        }
    } else {
        for (int idx = lane; idx < 16 * C; idx += 32) {
            int col = idx & (C - 1), row = idx >> 8;
            float y = (sOut[wv][row][col] - sMean[wv][row]) * sRstd[wv][row] * gamma[col] + beta[col];
            out[((size_t)(b * NPIX + m0 + row)) * C + col] = (_Float16)y;
        }
    }
}

// ---------------------------------------------------------------------------
// Kernel 3: S = q @ k^T.  q16 (B,L,C), k16T (B,C,N) -> S (B,L,N) f32.
// One wave: 16 L-rows x 256 pixels (16 WMMA tiles), A frags resident.
// ---------------------------------------------------------------------------
__global__ void __launch_bounds__(128)
attn_logits_kernel(const _Float16* __restrict__ q16, const _Float16* __restrict__ k16T,
                   float* __restrict__ S) {
    const int lane = threadIdx.x & 31;
    const int wid  = (blockIdx.x * blockDim.x + threadIdx.x) >> 5;
    const int NCH  = NPIX / 256;                 // 100 chunks of 256 pixels
    const int b  = wid / ((L / 16) * NCH);
    const int r  = wid % ((L / 16) * NCH);
    const int mt = r / NCH, nc = r % NCH;
    const int m0 = mt * 16;

    const int kb8 = (lane < 16) ? 0 : 8;
    const _Float16* qr = q16 + ((size_t)b * L + m0 + (lane & 15)) * C;
    v16h A[8];
    #pragma unroll
    for (int t = 0; t < 8; ++t) {
        h8 lo = *(const h8*)(qr + t * 32 + kb8);
        h8 hi = *(const h8*)(qr + t * 32 + kb8 + 16);
        A[t] = __builtin_shufflevector(lo, hi, 0, 1, 2, 3, 4, 5, 6, 7,
                                               8, 9, 10, 11, 12, 13, 14, 15);
    }

    const _Float16* kb_ = k16T + (size_t)b * C * NPIX;
    float* Sb = S + (size_t)b * L * NPIX;
    const int krow = (lane & 15) + ((lane < 16) ? 0 : 16);
    const int nn15 = lane & 15;
    const int mof  = (lane < 16) ? 0 : 8;

    for (int nt = 0; nt < 16; ++nt) {
        int n0 = nc * 256 + nt * 16;
        v8f acc = {};
        #pragma unroll
        for (int t = 0; t < 8; ++t) {
            v16h Bf;
            #pragma unroll
            for (int i = 0; i < 8; ++i) {
                Bf[i]     = kb_[(size_t)(t * 32 + kb8 + i) * NPIX + n0 + nn15];
                Bf[8 + i] = kb_[(size_t)(t * 32 + kb8 + 16 + i) * NPIX + n0 + nn15];
            }
            acc = wmma_f16(A[t], Bf, acc);
        }
        #pragma unroll
        for (int rr = 0; rr < 8; ++rr)
            Sb[(size_t)(m0 + mof + rr) * NPIX + n0 + nn15] = acc[rr];
    }
}

// ---------------------------------------------------------------------------
// Kernel 4: column softmax over L (axis=1) + mask zero-fill, in place.
// One thread per (b, pixel); strided column walks coalesce across lanes.
// ---------------------------------------------------------------------------
__global__ void __launch_bounds__(256)
softmax_mask_kernel(float* __restrict__ S, const unsigned char* __restrict__ mask) {
    int idx = blockIdx.x * blockDim.x + threadIdx.x;
    if (idx >= Bb * NPIX) return;
    int b = idx / NPIX, n = idx - b * NPIX;
    float* col = S + (size_t)b * L * NPIX + n;
    float mx = -3.4e38f;
    for (int l = 0; l < L; ++l) mx = fmaxf(mx, col[(size_t)l * NPIX]);
    float sum = 0.f;
    for (int l = 0; l < L; ++l) sum += __expf(col[(size_t)l * NPIX] - mx);
    float scale = (mask[n] ? 1.0f : 0.0f) / sum;
    for (int l = 0; l < L; ++l)
        col[(size_t)l * NPIX] = __expf(col[(size_t)l * NPIX] - mx) * scale;
}

// ---------------------------------------------------------------------------
// Kernel 5: O = S @ v.  S (B,L,N) f32 (A, cvt f16), v16 (B,N,C) f16 (B) -> f32.
// One wave per 16x16 output tile, K = 25600 in steps of 32.
// ---------------------------------------------------------------------------
__global__ void __launch_bounds__(128)
attn_out_kernel(const float* __restrict__ S, const _Float16* __restrict__ v16p,
                float* __restrict__ O) {
    const int lane = threadIdx.x & 31;
    const int wid  = (blockIdx.x * blockDim.x + threadIdx.x) >> 5;
    const int b  = wid / ((L / 16) * (C / 16));
    const int r  = wid % ((L / 16) * (C / 16));
    const int mt = r / (C / 16), ct = r % (C / 16);
    const int m0 = mt * 16, c0 = ct * 16;

    const int kb8 = (lane < 16) ? 0 : 8;
    const float* Sr = S + (size_t)b * L * NPIX + (size_t)(m0 + (lane & 15)) * NPIX;
    const _Float16* vb = v16p + (size_t)b * NPIX * C;
    const int klane = (lane & 15) + ((lane < 16) ? 0 : 16);

    v8f acc = {};
    for (int k0 = 0; k0 < NPIX; k0 += 32) {
        v16h Af;
        const float* p0 = Sr + k0 + kb8;
        const float* p1 = p0 + 16;
        #pragma unroll
        for (int i = 0; i < 8; ++i) {
            Af[i]     = (_Float16)p0[i];
            Af[8 + i] = (_Float16)p1[i];
        }
        v16h Bf;
        #pragma unroll
        for (int i = 0; i < 8; ++i) {
            Bf[i]     = vb[(size_t)(k0 + kb8 + i) * C + c0 + (lane & 15)];
            Bf[8 + i] = vb[(size_t)(k0 + kb8 + 16 + i) * C + c0 + (lane & 15)];
        }
        acc = wmma_f16(Af, Bf, acc);
    }
    const int nn15 = lane & 15;
    const int mof  = (lane < 16) ? 0 : 8;
    #pragma unroll
    for (int rr = 0; rr < 8; ++rr)
        O[((size_t)(b * L + m0 + mof + rr)) * C + c0 + nn15] = acc[rr];
}

// ---------------------------------------------------------------------------
// Kernel 6: out = relu(LN(O)).  One wave per row, wave32 shuffle reductions.
// ---------------------------------------------------------------------------
__global__ void __launch_bounds__(128)
final_ln_kernel(const float* __restrict__ O, const float* __restrict__ g1,
                const float* __restrict__ bt1, float* __restrict__ out) {
    const int row  = (blockIdx.x * blockDim.x + threadIdx.x) >> 5;
    const int lane = threadIdx.x & 31;
    if (row >= Bb * L) return;
    const float* x = O + (size_t)row * C;

    float s = 0.f;
    for (int i = lane; i < C; i += 32) s += x[i];
    #pragma unroll
    for (int off = 16; off > 0; off >>= 1) s += __shfl_xor(s, off, 32);
    float m = s * (1.0f / C);

    float vv = 0.f;
    for (int i = lane; i < C; i += 32) { float d = x[i] - m; vv += d * d; }
    #pragma unroll
    for (int off = 16; off > 0; off >>= 1) vv += __shfl_xor(vv, off, 32);
    float rstd = rsqrtf(vv * (1.0f / C) + EPS);

    for (int i = lane; i < C; i += 32) {
        float y = (x[i] - m) * rstd * g1[i] + bt1[i];
        out[(size_t)row * C + i] = fmaxf(y, 0.f);
    }
}

// ---------------------------------------------------------------------------
// Workspace layout (bytes).  Total ~159.3 MB.
// ---------------------------------------------------------------------------
constexpr size_t O_WQT = 0;
constexpr size_t O_WKT = O_WQT + (size_t)C * C * 2;
constexpr size_t O_WVT = O_WKT + (size_t)C * C * 2;
constexpr size_t O_Q16 = O_WVT + (size_t)C * C * 2;
constexpr size_t O_K16 = O_Q16 + (size_t)Bb * L * C * 2;
constexpr size_t O_V16 = O_K16 + (size_t)Bb * C * NPIX * 2;
constexpr size_t O_S   = O_V16 + (size_t)Bb * NPIX * C * 2;
constexpr size_t O_OUT = O_S   + (size_t)Bb * L * NPIX * 4;

extern "C" void kernel_launch(void* const* d_in, const int* in_sizes, int n_in,
                              void* d_out, int out_size, void* d_ws, size_t ws_size,
                              hipStream_t stream) {
    const float* pro  = (const float*)d_in[0];
    const float* feat = (const float*)d_in[1];
    const float* pos  = (const float*)d_in[2];
    const unsigned char* mask = (const unsigned char*)d_in[3];
    const float* Wq = (const float*)d_in[4];
    const float* bq = (const float*)d_in[5];
    const float* Wk = (const float*)d_in[6];
    const float* bk = (const float*)d_in[7];
    const float* Wv = (const float*)d_in[8];
    const float* bv = (const float*)d_in[9];
    const float* gq  = (const float*)d_in[10];
    const float* btq = (const float*)d_in[11];
    const float* gk  = (const float*)d_in[12];
    const float* btk = (const float*)d_in[13];
    const float* gv  = (const float*)d_in[14];
    const float* btv = (const float*)d_in[15];
    const float* g1  = (const float*)d_in[16];
    const float* bt1 = (const float*)d_in[17];
    float* out = (float*)d_out;

    char* ws = (char*)d_ws;
    _Float16* WqT = (_Float16*)(ws + O_WQT);
    _Float16* WkT = (_Float16*)(ws + O_WKT);
    _Float16* WvT = (_Float16*)(ws + O_WVT);
    _Float16* q16 = (_Float16*)(ws + O_Q16);
    _Float16* k16 = (_Float16*)(ws + O_K16);
    _Float16* v16 = (_Float16*)(ws + O_V16);
    float*    S   = (float*)(ws + O_S);
    float*    O   = (float*)(ws + O_OUT);

    // 0) weight transposes -> f16
    wt_kernel<<<(C * C) / 256, 256, 0, stream>>>(Wq, WqT);
    wt_kernel<<<(C * C) / 256, 256, 0, stream>>>(Wk, WkT);
    wt_kernel<<<(C * C) / 256, 256, 0, stream>>>(Wv, WvT);

    // 1) q projection + LN
    proj_q_kernel<<<Bb * (L / 32), 64, 0, stream>>>(pro, WqT, bq, gq, btq, q16);

    // 2) k (pos added, transposed out) and v projections + LN
    proj_feat_kernel<<<Bb * (NPIX / 32), 64, 0, stream>>>(feat, pos, WkT, bk, gk, btk,
                                                          k16, /*addpos=*/1, /*transposed=*/1);
    proj_feat_kernel<<<Bb * (NPIX / 32), 64, 0, stream>>>(feat, pos, WvT, bv, gv, btv,
                                                          v16, /*addpos=*/0, /*transposed=*/0);

    // 3) logits S = q @ k^T
    {
        int waves = Bb * (L / 16) * (NPIX / 256);   // 6400
        attn_logits_kernel<<<waves / 4, 128, 0, stream>>>(q16, k16, S);
    }

    // 4) softmax over L + mask
    {
        int nthreads = Bb * NPIX;                    // 51200
        softmax_mask_kernel<<<(nthreads + 255) / 256, 256, 0, stream>>>(S, mask);
    }

    // 5) O = S @ v
    {
        int waves = Bb * (L / 16) * (C / 16);        // 1024
        attn_out_kernel<<<waves / 4, 128, 0, stream>>>(S, v16, O);
    }

    // 6) final LN + ReLU
    {
        int waves = Bb * L;                          // 1024
        final_ln_kernel<<<waves / 4, 128, 0, stream>>>(O, g1, bt1, out);
    }
}
